// InteractionPharmacoPhoreEncoder_10058813407616
// MI455X (gfx1250) — matrix operand order repaired and checked by reading, not compiled, not run on hardware
//
#include <hip/hip_runtime.h>
#include <hip/hip_bf16.h>

// ---------------- problem constants (from reference setup) ----------------
#define BATCH 256
#define NI    50
#define PDIM  9
#define BP    256
#define NP    2048
#define NL    128
#define DDIM  256
#define MAXI  50
#define EPSV  1e-5f

#define MT    64            // padded M/N (50 -> 64, 4 tiles of 16)
#define LDA   264           // halves per LDS row (256 + 8 pad; 528B row, 16B aligned)
#define MMLD  65            // f32 stride for mm tile store (conflict-free-ish)

typedef __attribute__((ext_vector_type(16))) _Float16 v16h;
typedef __attribute__((ext_vector_type(8)))  _Float16 v8h;
typedef __attribute__((ext_vector_type(8)))  float    v8f;
typedef __attribute__((ext_vector_type(4)))  float    v4f;

// PARAM_IDX = [[0],[0],[0],[0,8],[0,1],[0,4,5,7],[0,4,5],[0],[0,2,3]]
__device__ __constant__ int c_pidx[9][4] = {
    {0,0,0,0},{0,0,0,0},{0,0,0,0},{0,8,0,0},{0,1,0,0},
    {0,4,5,7},{0,4,5,0},{0,0,0,0},{0,2,3,0}};
__device__ __constant__ int c_plen[9] = {1,1,1,2,2,4,3,1,3};

__device__ __forceinline__ float sigmoidf_(float x) {
    return 1.0f / (1.0f + __expf(-x));
}

// ---------------- kernel 0: zero cmax ----------------
__global__ void k_init(int* cmax) {
    if (threadIdx.x < 9) cmax[threadIdx.x] = 0;
}

// ---------------- kernel 1: per-class max count across batch ----------------
__global__ void k_cmax(const int* __restrict__ inter_type, int* __restrict__ cmax) {
    int b = blockIdx.x;
    int c = threadIdx.x;
    if (c < 9) {
        int cnt = 0;
        for (int i = 0; i < NI; ++i) cnt += (inter_type[b * NI + i] == c) ? 1 : 0;
        atomicMax(&cmax[c], cnt);
    }
}

// ---------------- kernel 2: sasa sums + LayerNorm(4) -> out[:,0:4] ----------------
__global__ void k_sasa(const float* __restrict__ sasa, const float* __restrict__ g,
                       const float* __restrict__ bb, float* __restrict__ out) {
    __shared__ float s[4];
    int b = blockIdx.x;
    int t = threadIdx.x;
    if (t < 4) {
        float acc = 0.f;
        for (int i = 0; i < NI; ++i) acc += sasa[(b * NI + i) * 4 + t];
        s[t] = acc;
    }
    __syncthreads();
    if (t < 4) {
        float m = (s[0] + s[1] + s[2] + s[3]) * 0.25f;
        float v = 0.f;
        #pragma unroll
        for (int k = 0; k < 4; ++k) { float d = s[k] - m; v += d * d; }
        v *= 0.25f;
        out[b * 13 + t] = (s[t] - m) * rsqrtf(v + EPSV) * g[t] + bb[t];
    }
}

// ---------------- kernel 3: per (b,c) gather + WMMA GEMM + fused epilogue ----------------
__global__ __launch_bounds__(128) void k_main(
    const float* __restrict__ params, const float* __restrict__ h_prot,
    const float* __restrict__ h_ligand,
    const float* __restrict__ W1, const float* __restrict__ b1,
    const float* __restrict__ W2, const float* __restrict__ b2,
    const float* __restrict__ W3, const float* __restrict__ b3,
    const float* __restrict__ pn_g, const float* __restrict__ pn_b,
    const float* __restrict__ mm_g, const float* __restrict__ mm_b,
    const float* __restrict__ decay,
    const int* __restrict__ inter_type, const int* __restrict__ pairs_lig,
    const int* __restrict__ pairs_pro, const int* __restrict__ prot_idx,
    const int* __restrict__ cmax, float* __restrict__ out)
{
    extern __shared__ char smem[];
    _Float16* Ash = (_Float16*)smem;               // MT*LDA halves (protein rows, M x K)
    _Float16* Bsh = Ash + MT * LDA;                // MT*LDA halves (ligand rows,  N x K)
    int*      ord = (int*)(Bsh + MT * LDA);        // 64 ints
    float*  termS = (float*)(ord + 64);            // 64 floats
    int*     misc = (int*)(termS + 64);            // [0] = count
    float*    mmS = (float*)Ash;                   // overlay A region AFTER barrier (64*65 f32)

    int bc = blockIdx.x;
    int b  = bc / 9;
    int c  = bc % 9;
    int tid  = threadIdx.x;
    int w    = tid >> 5;
    int lane = tid & 31;

    // ---- stable compaction (masked positions first, original order) ----
    if (tid == 0) {
        int cnt = 0;
        for (int i = 0; i < NI; ++i)
            if (inter_type[b * NI + i] == c) ord[cnt++] = i;
        misc[0] = cnt;
    }
    __syncthreads();
    int count = misc[0];

    // ---- stage gathered A (pro_emb) and B (lig_emb) rows into LDS as f16 ----
    int pb = prot_idx[b];
    for (int r = w; r < MT; r += 4) {
        int col = lane * 8;
        v8h av, bv;
        if (r < count) {
            int o  = ord[r];
            const float* srcA = h_prot   + ((size_t)pb * NP + (size_t)pairs_pro[b * NI + o]) * DDIM;
            const float* srcB = h_ligand + ((size_t)b  * NL + (size_t)pairs_lig[b * NI + o]) * DDIM;
            v4f a0 = *(const v4f*)(srcA + col);
            v4f a1 = *(const v4f*)(srcA + col + 4);
            v4f b0 = *(const v4f*)(srcB + col);
            v4f b1v = *(const v4f*)(srcB + col + 4);
            #pragma unroll
            for (int j = 0; j < 4; ++j) {
                av[j]     = (_Float16)a0[j];
                av[j + 4] = (_Float16)a1[j];
                bv[j]     = (_Float16)b0[j];
                bv[j + 4] = (_Float16)b1v[j];
            }
        } else {
            #pragma unroll
            for (int j = 0; j < 8; ++j) { av[j] = (_Float16)0.f; bv[j] = (_Float16)0.f; }
        }
        *(v8h*)&Ash[r * LDA + col] = av;
        *(v8h*)&Bsh[r * LDA + col] = bv;
    }
    __syncthreads();

    // ---- WMMA: wave w owns M-tile w; accumulate 4 N-tiles over K=256 ----
    v8f acc[4];
    #pragma unroll
    for (int nt = 0; nt < 4; ++nt)
        #pragma unroll
        for (int r = 0; r < 8; ++r) acc[nt][r] = 0.f;

    int m0 = w * 16 + (lane & 15);
    int hh = lane >> 4;

    #pragma unroll
    for (int ks = 0; ks < 8; ++ks) {
        int k0 = ks * 32;
        // A 16x32 f16 layout: lane (m = lane&15, h = lane>>4): v0-3 = K 8h+0..7, v4-7 = K 16+8h+0..7
        v8h alo = *(const v8h*)&Ash[m0 * LDA + k0 + 8 * hh];
        v8h ahi = *(const v8h*)&Ash[m0 * LDA + k0 + 16 + 8 * hh];
        v16h afrag;
        #pragma unroll
        for (int j = 0; j < 8; ++j) { afrag[j] = alo[j]; afrag[j + 8] = ahi[j]; }

        #pragma unroll
        for (int nt = 0; nt < 4; ++nt) {
            // B 32x16 f16 layout: lane (n = lane&15, h): v0-3 = K 8h+0..7, v4-7 = K 16+8h+0..7 of column n
            int n  = nt * 16 + (lane & 15);
            int kb = k0 + 8 * hh;
            v8h blo = *(const v8h*)&Bsh[n * LDA + kb];
            v8h bhi = *(const v8h*)&Bsh[n * LDA + kb + 16];
            v16h bfrag;
            #pragma unroll
            for (int j = 0; j < 8; ++j) { bfrag[j] = blo[j]; bfrag[j + 8] = bhi[j]; }
            acc[nt] = __builtin_amdgcn_wmma_f32_16x16x32_f16(
                false, afrag, false, bfrag, (short)0, acc[nt], false, false);
        }
    }
    __syncthreads();   // all LDS A/B reads complete before overlaying mmS on Ash

    // ---- spill C tiles: VGPR r -> (M = mt*16 + r + 8h, N = nt*16 + lane&15) ----
    #pragma unroll
    for (int nt = 0; nt < 4; ++nt)
        #pragma unroll
        for (int r = 0; r < 8; ++r) {
            int row = w * 16 + r + 8 * hh;
            int col = nt * 16 + (lane & 15);
            mmS[row * MMLD + col] = acc[nt][r];
        }
    __syncthreads();

    // ---- epilogue: one thread per interaction row i < 50 ----
    if (tid < NI) {
        int i = tid;
        // LayerNorm over mm row (50 elems) then dot W2 + sigmoid
        float s = 0.f, ss = 0.f;
        for (int j = 0; j < NI; ++j) {
            float x = mmS[i * MMLD + j];
            s += x; ss += x * x;
        }
        float m   = s * (1.0f / NI);
        float var = ss * (1.0f / NI) - m * m;
        float inv = rsqrtf(var + EPSV);
        float dot = 0.f;
        for (int j = 0; j < NI; ++j) {
            float x = (mmS[i * MMLD + j] - m) * inv * mm_g[c * MAXI + j] + mm_b[c * MAXI + j];
            dot += x * W2[c * MAXI + j];
        }
        float s2 = sigmoidf_(dot + b2[c]);

        // param LN over k selected columns, dot W3
        int   k     = c_plen[c];
        float valid = (i < count) ? 1.f : 0.f;
        const float* prow = params + ((size_t)b * NI + (size_t)((i < count) ? ord[i] : 0)) * PDIM;
        float vals[4];
        #pragma unroll
        for (int kk = 0; kk < 4; ++kk)
            vals[kk] = (kk < k) ? prow[c_pidx[c][kk]] * valid : 0.f;
        float pm = 0.f;
        for (int kk = 0; kk < k; ++kk) pm += vals[kk];
        pm /= (float)k;
        float pv = 0.f;
        for (int kk = 0; kk < k; ++kk) { float d = vals[kk] - pm; pv += d * d; }
        pv /= (float)k;
        float pinv = rsqrtf(pv + EPSV);
        float p3 = b3[c];
        for (int kk = 0; kk < k; ++kk)
            p3 += ((vals[kk] - pm) * pinv * pn_g[c * 4 + kk] + pn_b[c * 4 + kk]) * W3[c * 4 + kk];

        float pg0 = prow[0] * valid;
        float pad = (i < cmax[c]) ? 1.f : 0.f;
        float pa  = sigmoidf_(__expf(-decay[c] * pg0) * p3) * pad;
        termS[i] = s2 * pa * W1[c * MAXI + i];
    }
    __syncthreads();
    if (tid == 0) {
        float accv = b1[c];
        for (int i = 0; i < NI; ++i) accv += termS[i];
        out[b * 13 + 4 + c] = accv;
    }
}

// ---------------- host launcher ----------------
extern "C" void kernel_launch(void* const* d_in, const int* in_sizes, int n_in,
                              void* d_out, int out_size, void* d_ws, size_t ws_size,
                              hipStream_t stream) {
    (void)in_sizes; (void)n_in; (void)out_size; (void)ws_size;
    const float* params   = (const float*)d_in[0];
    const float* h_prot   = (const float*)d_in[1];
    const float* h_ligand = (const float*)d_in[2];
    const float* sasa     = (const float*)d_in[3];
    const float* W1   = (const float*)d_in[4];
    const float* b1   = (const float*)d_in[5];
    const float* W2   = (const float*)d_in[6];
    const float* b2   = (const float*)d_in[7];
    const float* W3   = (const float*)d_in[8];
    const float* b3   = (const float*)d_in[9];
    const float* pn_g = (const float*)d_in[10];
    const float* pn_b = (const float*)d_in[11];
    const float* mm_g = (const float*)d_in[12];
    const float* mm_b = (const float*)d_in[13];
    const float* dec  = (const float*)d_in[14];
    const float* ln4g = (const float*)d_in[15];
    const float* ln4b = (const float*)d_in[16];
    const int* inter_type = (const int*)d_in[17];
    const int* pairs_lig  = (const int*)d_in[18];
    const int* pairs_pro  = (const int*)d_in[19];
    const int* prot_idx   = (const int*)d_in[20];
    float* out = (float*)d_out;
    int*  cmax = (int*)d_ws;

    k_init<<<1, 32, 0, stream>>>(cmax);
    k_cmax<<<BATCH, 32, 0, stream>>>(inter_type, cmax);
    k_sasa<<<BATCH, 32, 0, stream>>>(sasa, ln4g, ln4b, out);

    size_t smem = (size_t)MT * LDA * 2 * sizeof(_Float16)  // A + B f16 stages
                + 64 * sizeof(int)                         // ord
                + 64 * sizeof(float)                       // termS
                + 16;                                      // misc
    k_main<<<BATCH * 9, 128, smem, stream>>>(
        params, h_prot, h_ligand, W1, b1, W2, b2, W3, b3,
        pn_g, pn_b, mm_g, mm_b, dec,
        inter_type, pairs_lig, pairs_pro, prot_idx, cmax, out);
}
